// Attention_13039520711118
// MI455X (gfx1250) — matrix operand order repaired and checked by reading, not compiled, not run on hardware
//
#include <hip/hip_runtime.h>
#include <hip/hip_bf16.h>
#include <math.h>

typedef __attribute__((ext_vector_type(16))) __bf16 v16bf;
typedef __attribute__((ext_vector_type(8)))  float  v8f;

#define NEG_INF_F (-1000000000.0f)
constexpr int B_ = 16, S_ = 2048, E_ = 2048, H_ = 1024;
constexpr int MW_STRIDE = E_ + H_; // 3072

// ---------------- Stage 1a: M_w[:, :E] -> bf16 workspace (row-major H x E) ---------
__global__ void cvt_mw_kernel(const float* __restrict__ Mw, __bf16* __restrict__ out) {
    int idx = blockIdx.x * 256 + threadIdx.x;       // 0 .. H*E-1
    int h = idx >> 11;                              // E_ == 2048
    int e = idx & (E_ - 1);
    out[idx] = (__bf16)Mw[(size_t)h * MW_STRIDE + e];
}

// ---------------- Stage 1b: hb[b,h] = M_b[h] + sum_d hidden[b,d]*M_w[h,E+d] --------
__global__ void hproj_kernel(const float* __restrict__ Mw, const float* __restrict__ Mb,
                             const float* __restrict__ hidden, float* __restrict__ hb) {
    int h = blockIdx.x * 256 + threadIdx.x;
    int b = blockIdx.y;
    const float* hv = hidden + (size_t)b * H_;
    const float* mr = Mw + (size_t)h * MW_STRIDE + E_;
    float s = Mb[h];
    for (int d = 0; d < H_; ++d) s = fmaf(hv[d], mr[d], s);
    hb[b * H_ + h] = s;
}

// ---------------- Stage 2: scores via bf16 WMMA -----------------------------------
// Block: 256 threads = 8 waves. Grid: (S/16, B). Each wave owns 8 h-tiles of 16.
__global__ void __launch_bounds__(256) attn_scores_kernel(
    const float* __restrict__ enc, const __bf16* __restrict__ mwbf,
    const float* __restrict__ hb, const float* __restrict__ Vw,
    const float* __restrict__ Vb, const unsigned char* __restrict__ mask,
    float* __restrict__ scores) {
    const int stile  = blockIdx.x;           // 0..127
    const int b      = blockIdx.y;           // 0..15
    const int tid    = threadIdx.x;
    const int wave   = tid >> 5;
    const int lane   = tid & 31;
    const int l15    = lane & 15;
    const int hiHalf = lane >> 4;

    __shared__ float ssum[16];
    if (tid < 16) ssum[tid] = 0.0f;
    __syncthreads();

    // A: encoded rows (M = l15 for both lane halves), K contiguous per half:
    //   element i -> K = kbase+i, kbase = 16*hiHalf
    const float* arow = enc + (((size_t)b * S_) + (size_t)stile * 16 + l15) * E_;
    const int kbase = hiHalf * 16;

    // B: Mw^T tiles. Lane = column h_local (l15), K contiguous per half:
    //   element i -> K = 16*hiHalf + i  (same contiguous order as A)
    const __bf16* brow = mwbf + ((size_t)(wave * 8 * 16 + l15)) * E_ + 16 * hiHalf;

    v8f acc[8];
    v8f zero = {0.f, 0.f, 0.f, 0.f, 0.f, 0.f, 0.f, 0.f};
#pragma unroll
    for (int t = 0; t < 8; ++t) acc[t] = zero;

    for (int k0 = 0; k0 < E_; k0 += 32) {
        const float4* a0 = reinterpret_cast<const float4*>(arow + k0 + kbase);
        float4 f0 = a0[0], f1 = a0[1];
        const float4* a1 = reinterpret_cast<const float4*>(arow + k0 + kbase + 8);
        float4 f2 = a1[0], f3 = a1[1];
        v16bf afrag;
        afrag[0]  = (__bf16)f0.x; afrag[1]  = (__bf16)f0.y;
        afrag[2]  = (__bf16)f0.z; afrag[3]  = (__bf16)f0.w;
        afrag[4]  = (__bf16)f1.x; afrag[5]  = (__bf16)f1.y;
        afrag[6]  = (__bf16)f1.z; afrag[7]  = (__bf16)f1.w;
        afrag[8]  = (__bf16)f2.x; afrag[9]  = (__bf16)f2.y;
        afrag[10] = (__bf16)f2.z; afrag[11] = (__bf16)f2.w;
        afrag[12] = (__bf16)f3.x; afrag[13] = (__bf16)f3.y;
        afrag[14] = (__bf16)f3.z; afrag[15] = (__bf16)f3.w;
#pragma unroll
        for (int t = 0; t < 8; ++t) {
            v16bf bfrag = *reinterpret_cast<const v16bf*>(brow + (size_t)t * 16 * E_ + k0);
            acc[t] = __builtin_amdgcn_wmma_f32_16x16x32_bf16(
                false, afrag, false, bfrag, (short)0, acc[t], false, false);
        }
    }

    // Epilogue: D layout -> lane column h_local = l15, rows M = r + 8*hiHalf.
    float sumr[8];
#pragma unroll
    for (int r = 0; r < 8; ++r) sumr[r] = 0.0f;
#pragma unroll
    for (int t = 0; t < 8; ++t) {
        int h = (wave * 8 + t) * 16 + l15;
        float hbv = hb[b * H_ + h];
        float vwv = Vw[h];
#pragma unroll
        for (int r = 0; r < 8; ++r)
            sumr[r] += tanhf(acc[t][r] + hbv) * vwv;
    }
    // Reduce across the 16 lanes of each half (xor 1,2,4,8 stays within a half).
#pragma unroll
    for (int r = 0; r < 8; ++r) {
        float v = sumr[r];
        v += __shfl_xor(v, 1);
        v += __shfl_xor(v, 2);
        v += __shfl_xor(v, 4);
        v += __shfl_xor(v, 8);
        sumr[r] = v;
    }
    if (l15 == 0) {
#pragma unroll
        for (int r = 0; r < 8; ++r) atomicAdd(&ssum[r + 8 * hiHalf], sumr[r]);
    }
    __syncthreads();
    if (tid < 16) {
        int s = stile * 16 + tid;
        float sc = ssum[tid] + Vb[0];
        if (mask[(size_t)b * S_ + s]) sc = NEG_INF_F;
        scores[(size_t)b * S_ + s] = sc;
    }
}

// ---------------- Stage 3: masked softmax per batch row ---------------------------
__global__ void __launch_bounds__(256) softmax_kernel(const float* __restrict__ scores,
                                                      float* __restrict__ weights) {
    const int b = blockIdx.x, tid = threadIdx.x;
    __shared__ float red[8];
    __shared__ float bcast;
    const float* sc = scores + (size_t)b * S_;
    float v[8];
    float mx = -3.4e38f;
#pragma unroll
    for (int i = 0; i < 8; ++i) { v[i] = sc[tid + i * 256]; mx = fmaxf(mx, v[i]); }
#pragma unroll
    for (int off = 16; off >= 1; off >>= 1) mx = fmaxf(mx, __shfl_xor(mx, off));
    if ((tid & 31) == 0) red[tid >> 5] = mx;
    __syncthreads();
    if (tid == 0) {
        float m = red[0];
        for (int i = 1; i < 8; ++i) m = fmaxf(m, red[i]);
        bcast = m;
    }
    __syncthreads();
    mx = bcast;
    float sum = 0.0f;
#pragma unroll
    for (int i = 0; i < 8; ++i) { v[i] = __expf(v[i] - mx); sum += v[i]; }
#pragma unroll
    for (int off = 16; off >= 1; off >>= 1) sum += __shfl_xor(sum, off);
    if ((tid & 31) == 0) red[tid >> 5] = sum;
    __syncthreads();
    if (tid == 0) {
        float s = 0.0f;
        for (int i = 0; i < 8; ++i) s += red[i];
        bcast = 1.0f / s;
    }
    __syncthreads();
    float inv = bcast;
#pragma unroll
    for (int i = 0; i < 8; ++i) weights[(size_t)b * S_ + tid + i * 256] = v[i] * inv;
}

// ---------------- Stage 4: weighted[b,e] = sum_s w[b,s]*enc[b,s,e] ----------------
__global__ void __launch_bounds__(256) weighted_kernel(const float* __restrict__ enc,
                                                       const float* __restrict__ w,
                                                       float* __restrict__ out) {
    const int b = blockIdx.y;
    const int e = blockIdx.x * 256 + threadIdx.x;
    const float* wp = w + (size_t)b * S_;
    const float* ep = enc + ((size_t)b * S_) * E_ + e;
    float a0 = 0.f, a1 = 0.f, a2 = 0.f, a3 = 0.f;
    for (int s = 0; s < S_; s += 4) {
        a0 = fmaf(wp[s + 0], ep[(size_t)(s + 0) * E_], a0);
        a1 = fmaf(wp[s + 1], ep[(size_t)(s + 1) * E_], a1);
        a2 = fmaf(wp[s + 2], ep[(size_t)(s + 2) * E_], a2);
        a3 = fmaf(wp[s + 3], ep[(size_t)(s + 3) * E_], a3);
    }
    out[(size_t)b * E_ + e] = (a0 + a1) + (a2 + a3);
}

extern "C" void kernel_launch(void* const* d_in, const int* in_sizes, int n_in,
                              void* d_out, int out_size, void* d_ws, size_t ws_size,
                              hipStream_t stream) {
    const float* enc            = (const float*)d_in[0];            // (B,S,E)
    const float* hidden         = (const float*)d_in[1];            // (B,1,H)
    const unsigned char* mask   = (const unsigned char*)d_in[2];    // (B,S) bool (1 byte)
    const float* Mw             = (const float*)d_in[3];            // (H, E+H)
    const float* Mb             = (const float*)d_in[4];            // (H,)
    const float* Vw             = (const float*)d_in[5];            // (1,H)
    const float* Vb             = (const float*)d_in[6];            // (1,)

    float* out          = (float*)d_out;
    float* weighted_out = out;              // (B,1,E) flat = B*E
    float* weights_out  = out + B_ * S_;    // (B,1,S) flat = B*S

    // Workspace layout: [Mw bf16: H*E*2 B][hb: B*H*4 B][scores: B*S*4 B] ~= 4.19 MiB
    char*   ws     = (char*)d_ws;
    __bf16* mwbf   = (__bf16*)ws;
    float*  hb     = (float*)(ws + (size_t)H_ * E_ * 2);
    float*  scores = (float*)(ws + (size_t)H_ * E_ * 2 + (size_t)B_ * H_ * 4);

    cvt_mw_kernel  <<<(H_ * E_) / 256, 256, 0, stream>>>(Mw, mwbf);
    hproj_kernel   <<<dim3(H_ / 256, B_), 256, 0, stream>>>(Mw, Mb, hidden, hb);
    attn_scores_kernel<<<dim3(S_ / 16, B_), 256, 0, stream>>>(enc, mwbf, hb, Vw, Vb, mask, scores);
    softmax_kernel <<<B_, 256, 0, stream>>>(scores, weights_out);
    weighted_kernel<<<dim3(E_ / 256, B_), 256, 0, stream>>>(enc, weights_out, weighted_out);
}
